// LSTM_CRF_52364241273406
// MI455X (gfx1250) — matrix operand order repaired and checked by reading, not compiled, not run on hardware
//
#include <hip/hip_runtime.h>
#include <hip/hip_bf16.h>

typedef __attribute__((ext_vector_type(16))) _Float16 v16h;
typedef __attribute__((ext_vector_type(8)))  _Float16 v8h;
typedef __attribute__((ext_vector_type(8)))  float    v8f;

#define T_LEN 1024
#define BATCH 256
#define IN_DIM 66
#define KX 96            // IN_DIM padded to multiple of 32
#define H_DIM 256        // per-direction hidden
#define G4 1024          // 4*H gates
#define TAGS 10
#define START_TAG 8
#define STOP_TAG 9
#define NEGV -10000.0f

#define NBLK 64          // persistent grid
#define SW_HH 272        // LDS row stride (f16) for W_hh slab: 136 dwords -> lane spacing 8 banks
#define SW_IH 104        // LDS row stride (f16) for W_ih slab: 52 dwords -> distinct bank windows

// ---------- WMMA helpers ----------
// A operand (16x32 f16): lane L<16 holds row L&15, K {kb..kb+7} u {kb+16..kb+23}, kb=(L>>4)*8
__device__ __forceinline__ v16h ldA(const _Float16* __restrict__ p) {
    v8h lo = *(const v8h*)(p);
    v8h hi = *(const v8h*)(p + 16);
    return __builtin_shufflevector(lo, hi, 0,1,2,3,4,5,6,7,8,9,10,11,12,13,14,15);
}
// B operand (32x16 f16): lane holds column n=lane&15, K {kb..kb+7} u {kb+16..kb+23}, kb=(lane>>4)*8
__device__ __forceinline__ v16h ldB(const _Float16* p) {
    v8h lo = *(const v8h*)(p);
    v8h hi = *(const v8h*)(p + 16);
    return __builtin_shufflevector(lo, hi, 0,1,2,3,4,5,6,7,8,9,10,11,12,13,14,15);
}
__device__ __forceinline__ v8f wmma(v16h a, v16h b, v8f c) {
    return __builtin_amdgcn_wmma_f32_16x16x32_f16(false, a, false, b, (short)0, c, false, false);
}
__device__ __forceinline__ float sigf(float x) { return 1.0f / (1.0f + expf(-x)); }

// ---------- grid-wide split barrier (64 co-resident blocks) ----------
__device__ __forceinline__ void grid_barrier(int* cnt, int* gen) {
    __syncthreads();
    __threadfence();                                   // release: h stores visible device-wide
    if (threadIdx.x == 0) {
        int g = __hip_atomic_load(gen, __ATOMIC_RELAXED, __HIP_MEMORY_SCOPE_AGENT);
        if (atomicAdd(cnt, 1) == NBLK - 1) {
            __hip_atomic_store(cnt, 0, __ATOMIC_RELAXED, __HIP_MEMORY_SCOPE_AGENT);
            __hip_atomic_fetch_add(gen, 1, __ATOMIC_RELEASE, __HIP_MEMORY_SCOPE_AGENT);
        } else {
            while (__hip_atomic_load(gen, __ATOMIC_ACQUIRE, __HIP_MEMORY_SCOPE_AGENT) == g)
                __builtin_amdgcn_s_sleep(2);
        }
    }
    __syncthreads();
    __threadfence();                                   // acquire: drop stale cached h lines
}

// ---------- prep: f32 -> f16 with row/K padding ----------
__global__ void k_cvt(const float* __restrict__ src, _Float16* __restrict__ dst,
                      int rows, int srcRows, int srcK, int dstK) {
    long long idx = (long long)blockIdx.x * blockDim.x + threadIdx.x;
    long long n = (long long)rows * dstK;
    if (idx >= n) return;
    int r = (int)(idx / dstK), k = (int)(idx % dstK);
    float v = (r < srcRows && k < srcK) ? src[(long long)r * srcK + k] : 0.0f;
    dst[idx] = (_Float16)v;
}

__global__ void k_zero_i(int* __restrict__ p, int n) {
    int i = blockIdx.x * blockDim.x + threadIdx.x;
    if (i < n) p[i] = 0;
}

// ---------- persistent BiLSTM: all 1024 steps, both directions ----------
// block -> (dir, jg, half); its 8 waves -> 8 batch tiles. W slabs LDS-stationary, c in VGPRs.
__global__ void __launch_bounds__(256)
k_lstm(const _Float16* __restrict__ x16,
       const _Float16* __restrict__ wih_f, const _Float16* __restrict__ whh_f, const float* __restrict__ bias_f,
       const _Float16* __restrict__ wih_b, const _Float16* __restrict__ whh_b, const float* __restrict__ bias_b,
       _Float16* __restrict__ hf_all, _Float16* __restrict__ hb_all,
       int* __restrict__ bar_cnt, int* __restrict__ bar_gen)
{
    __shared__ __align__(16) _Float16 sWhh[64 * SW_HH];   // 4 gates x 16 rows, K=256
    __shared__ __align__(16) _Float16 sWih[64 * SW_IH];   // 4 gates x 16 rows, K=96

    const int dir  = blockIdx.x >> 5;            // 0 fwd, 1 bwd
    const int jg   = (blockIdx.x >> 1) & 15;     // hidden-index tile (block-uniform)
    const int half = blockIdx.x & 1;
    const int lane = threadIdx.x & 31;
    const int btile = half * 8 + ((threadIdx.x >> 5) & 7);

    const _Float16* ghh  = dir ? whh_b  : whh_f;
    const _Float16* gih  = dir ? wih_b  : wih_f;
    const float*    bias = dir ? bias_b : bias_f;

    // ---- stage this block's weight slabs into LDS (once) ----
    for (int c = threadIdx.x; c < 64 * 32; c += 256) {           // 2048 x 16B chunks
        int row = c >> 5, off = (c & 31) * 8;
        int q = row >> 4, r = row & 15;
        *(v8h*)&sWhh[row * SW_HH + off] =
            *(const v8h*)(ghh + (size_t)(q * H_DIM + jg * 16 + r) * H_DIM + off);
    }
    for (int c = threadIdx.x; c < 64 * 12; c += 256) {           // 768 x 16B chunks
        int row = c / 12, off = (c % 12) * 8;
        int q = row >> 4, r = row & 15;
        *(v8h*)&sWih[row * SW_IH + off] =
            *(const v8h*)(gih + (size_t)(q * H_DIM + jg * 16 + r) * KX + off);
    }
    __syncthreads();

    const int arow = lane & 15;
    const int kb8  = (lane >> 4) * 8;
    const int kb16 = (lane >> 4) * 16;
    const int gb   = lane & 15;
    const int j    = jg * 16 + gb;

    const float bi = bias[0 * H_DIM + j];
    const float bf = bias[1 * H_DIM + j];
    const float bg = bias[2 * H_DIM + j];
    const float bo = bias[3 * H_DIM + j];

    float creg[8];                                // LSTM cell state lives in VGPRs
#pragma unroll
    for (int r = 0; r < 8; ++r) creg[r] = 0.0f;

    for (int t = 0; t < T_LEN; ++t) {
        if (t > 0) grid_barrier(bar_cnt, bar_gen);    // step t-1 h visible everywhere

        const int tt = dir ? (T_LEN - 1 - t) : t;
        const _Float16* xrow = x16 + ((size_t)tt * BATCH + btile * 16 + arow) * KX;
        if (t + 1 < T_LEN) {                          // x is read-only: prefetch next step
            const int tn = dir ? (T_LEN - 2 - t) : (t + 1);
            __builtin_prefetch((const void*)(x16 + ((size_t)tn * BATCH + btile * 16 + arow) * KX), 0, 1);
        }

        v8f acc0 = {}, acc1 = {}, acc2 = {}, acc3 = {};

        // input projection: K = 96, B from LDS
#pragma unroll
        for (int k0 = 0; k0 < KX; k0 += 32) {
            v16h a = ldA(xrow + k0 + kb8);
            acc0 = wmma(a, ldB(&sWih[(0 * 16 + gb) * SW_IH + k0 + kb8]), acc0);
            acc1 = wmma(a, ldB(&sWih[(1 * 16 + gb) * SW_IH + k0 + kb8]), acc1);
            acc2 = wmma(a, ldB(&sWih[(2 * 16 + gb) * SW_IH + k0 + kb8]), acc2);
            acc3 = wmma(a, ldB(&sWih[(3 * 16 + gb) * SW_IH + k0 + kb8]), acc3);
        }
        // recurrence: K = 256, B from LDS
        if (t > 0) {
            const _Float16* hrow = (dir ? hb_all + (size_t)(tt + 1) * BATCH * H_DIM
                                        : hf_all + (size_t)(t  - 1) * BATCH * H_DIM)
                                   + (size_t)(btile * 16 + arow) * H_DIM;
#pragma unroll
            for (int k0 = 0; k0 < H_DIM; k0 += 32) {
                v16h a = ldA(hrow + k0 + kb8);
                acc0 = wmma(a, ldB(&sWhh[(0 * 16 + gb) * SW_HH + k0 + kb8]), acc0);
                acc1 = wmma(a, ldB(&sWhh[(1 * 16 + gb) * SW_HH + k0 + kb8]), acc1);
                acc2 = wmma(a, ldB(&sWhh[(2 * 16 + gb) * SW_HH + k0 + kb8]), acc2);
                acc3 = wmma(a, ldB(&sWhh[(3 * 16 + gb) * SW_HH + k0 + kb8]), acc3);
            }
        }

        // LSTM cell in f32 (PyTorch gate order i, f, g, o); c stays in registers
        _Float16* hout = (dir ? hb_all : hf_all) + (size_t)tt * BATCH * H_DIM;
#pragma unroll
        for (int r = 0; r < 8; ++r) {
            const int row = btile * 16 + r + 8 * (lane >> 4);   // batch index
            const float iv = sigf(acc0[r] + bi);
            const float fv = sigf(acc1[r] + bf);
            const float gv = tanhf(acc2[r] + bg);
            const float ov = sigf(acc3[r] + bo);
            const float cn = fv * creg[r] + iv * gv;
            creg[r] = cn;
            hout[(size_t)row * H_DIM + j] = (_Float16)(ov * tanhf(cn));
        }
    }
}

// ---------- emissions: feats[T*B,16] = [h_f | h_b] @ lin_w^T + lin_b ----------
__global__ void __launch_bounds__(256)
k_feats(const _Float16* __restrict__ hf_all, const _Float16* __restrict__ hb_all,
        const _Float16* __restrict__ linw16, const float* __restrict__ lin_b,
        float* __restrict__ feats)
{
    const int wid  = (blockIdx.x * blockDim.x + threadIdx.x) >> 5;   // 0..16383
    const int lane = threadIdx.x & 31;
    const int m0 = wid * 16;             // global row base (t*B + b); tile stays in one t
    const int t  = m0 >> 8;
    const int b0 = m0 & 255;
    const int arow = lane & 15;
    const int kb8  = (lane >> 4) * 8;
    const int kb16 = (lane >> 4) * 16;
    const int col  = lane & 15;

    const _Float16* hf = hf_all + ((size_t)t * BATCH + b0 + arow) * H_DIM;
    const _Float16* hb = hb_all + ((size_t)t * BATCH + b0 + arow) * H_DIM;
    const _Float16* wr = linw16 + (size_t)col * (2 * H_DIM);

    v8f acc = {};
#pragma unroll
    for (int k0 = 0; k0 < H_DIM; k0 += 32)
        acc = wmma(ldA(hf + k0 + kb8), ldB(wr + k0 + kb8), acc);
#pragma unroll
    for (int k0 = 0; k0 < H_DIM; k0 += 32)
        acc = wmma(ldA(hb + k0 + kb8), ldB(wr + H_DIM + k0 + kb8), acc);

    const float bb = (col < TAGS) ? lin_b[col] : 0.0f;
#pragma unroll
    for (int r = 0; r < 8; ++r) {
        const int row = m0 + r + 8 * (lane >> 4);
        feats[(size_t)row * 16 + col] = acc[r] + bb;
    }
}

// ---------- CRF NLL per batch element ----------
__global__ void k_crf(const float* __restrict__ feats, const int* __restrict__ labels,
                      const float* __restrict__ transitions, float* __restrict__ nll_pb)
{
    __shared__ float tr[TAGS * TAGS];
    for (int i = threadIdx.x; i < TAGS * TAGS; i += blockDim.x) tr[i] = transitions[i];
    __syncthreads();

    const int b = blockIdx.x * blockDim.x + threadIdx.x;   // 0..255
    float alpha[TAGS];
#pragma unroll
    for (int tg = 0; tg < TAGS; ++tg) alpha[tg] = NEGV;
    alpha[START_TAG] = 0.0f;

    float gold_emit = 0.0f, gold_trans = 0.0f;
    int prev_lab = START_TAG;

    for (int t = 0; t < T_LEN; ++t) {
        float emitv[TAGS];
        const float* fr = feats + ((size_t)t * BATCH + b) * 16;
#pragma unroll
        for (int n = 0; n < TAGS; ++n) emitv[n] = fr[n];

        float na[TAGS];
#pragma unroll
        for (int n = 0; n < TAGS; ++n) {
            float m = -3.4e38f;
#pragma unroll
            for (int p = 0; p < TAGS; ++p) m = fmaxf(m, alpha[p] + tr[n * TAGS + p]);
            float s = 0.0f;
#pragma unroll
            for (int p = 0; p < TAGS; ++p) s += expf(alpha[p] + tr[n * TAGS + p] - m);
            na[n] = m + logf(s) + emitv[n];
        }
#pragma unroll
        for (int n = 0; n < TAGS; ++n) alpha[n] = na[n];

        const int lab = labels[(size_t)t * BATCH + b];
        gold_emit  += emitv[lab];
        gold_trans += tr[lab * TAGS + prev_lab];
        prev_lab = lab;
    }
    float gold = gold_trans + tr[STOP_TAG * TAGS + prev_lab] + gold_emit;

    float m = -3.4e38f;
#pragma unroll
    for (int tg = 0; tg < TAGS; ++tg) m = fmaxf(m, alpha[tg] + tr[STOP_TAG * TAGS + tg]);
    float s = 0.0f;
#pragma unroll
    for (int tg = 0; tg < TAGS; ++tg) s += expf(alpha[tg] + tr[STOP_TAG * TAGS + tg] - m);
    nll_pb[b] = (m + logf(s)) - gold;
}

// ---------- deterministic mean reduction ----------
__global__ void k_reduce(const float* __restrict__ pb, float* __restrict__ out) {
    __shared__ float s[BATCH];
    const int i = threadIdx.x;
    s[i] = pb[i];
    __syncthreads();
    for (int stride = BATCH / 2; stride > 0; stride >>= 1) {
        if (i < stride) s[i] += s[i + stride];
        __syncthreads();
    }
    if (i == 0) out[0] = s[0] / (float)BATCH;
}

extern "C" void kernel_launch(void* const* d_in, const int* in_sizes, int n_in,
                              void* d_out, int out_size, void* d_ws, size_t ws_size,
                              hipStream_t stream) {
    const float* features    = (const float*)d_in[0];
    const int*   labels      = (const int*)  d_in[1];
    /* lengths (d_in[2]) all == T_LEN -> identity */
    const float* w_ih_f      = (const float*)d_in[3];
    const float* w_hh_f      = (const float*)d_in[4];
    const float* b_f         = (const float*)d_in[5];
    const float* w_ih_b      = (const float*)d_in[6];
    const float* w_hh_b      = (const float*)d_in[7];
    const float* b_b         = (const float*)d_in[8];
    const float* lin_w       = (const float*)d_in[9];
    const float* lin_b       = (const float*)d_in[10];
    const float* transitions = (const float*)d_in[11];
    float* out = (float*)d_out;

    size_t off = 0;
    auto take = [&](size_t bytes) -> char* {
        char* p = (char*)d_ws + off;
        off += (bytes + 255) & ~(size_t)255;
        return p;
    };
    _Float16* x16     = (_Float16*)take((size_t)T_LEN * BATCH * KX * 2);
    _Float16* wihf16  = (_Float16*)take((size_t)G4 * KX * 2);
    _Float16* wihb16  = (_Float16*)take((size_t)G4 * KX * 2);
    _Float16* whhf16  = (_Float16*)take((size_t)G4 * H_DIM * 2);
    _Float16* whhb16  = (_Float16*)take((size_t)G4 * H_DIM * 2);
    _Float16* linw16  = (_Float16*)take((size_t)16 * 2 * H_DIM * 2);
    _Float16* hf_all  = (_Float16*)take((size_t)T_LEN * BATCH * H_DIM * 2);
    _Float16* hb_all  = (_Float16*)take((size_t)T_LEN * BATCH * H_DIM * 2);
    float*    feats   = (float*)   take((size_t)T_LEN * BATCH * 16 * 4);
    float*    nll_pb  = (float*)   take((size_t)BATCH * 4);
    int*      bar     = (int*)     take(2 * sizeof(int));   // [cnt, gen]

    auto cdiv = [](long long n, int d) { return (unsigned)((n + d - 1) / d); };

    // ---- convert inputs/weights to padded f16; init barrier ----
    {
        long long n = (long long)T_LEN * BATCH * KX;
        k_cvt<<<cdiv(n, 256), 256, 0, stream>>>(features, x16, T_LEN * BATCH, T_LEN * BATCH, IN_DIM, KX);
    }
    k_cvt<<<cdiv((long long)G4 * KX, 256), 256, 0, stream>>>(w_ih_f, wihf16, G4, G4, IN_DIM, KX);
    k_cvt<<<cdiv((long long)G4 * KX, 256), 256, 0, stream>>>(w_ih_b, wihb16, G4, G4, IN_DIM, KX);
    k_cvt<<<cdiv((long long)G4 * H_DIM, 256), 256, 0, stream>>>(w_hh_f, whhf16, G4, G4, H_DIM, H_DIM);
    k_cvt<<<cdiv((long long)G4 * H_DIM, 256), 256, 0, stream>>>(w_hh_b, whhb16, G4, G4, H_DIM, H_DIM);
    k_cvt<<<cdiv((long long)16 * 2 * H_DIM, 256), 256, 0, stream>>>(lin_w, linw16, 16, TAGS, 2 * H_DIM, 2 * H_DIM);
    k_zero_i<<<1, 64, 0, stream>>>(bar, 2);

    // ---- persistent BiLSTM over all timesteps (grid barrier between steps) ----
    k_lstm<<<NBLK, 256, 0, stream>>>(x16, wihf16, whhf16, b_f, wihb16, whhb16, b_b,
                                     hf_all, hb_all, bar, bar + 1);

    // ---- emissions for all timesteps (16384 WMMA tiles) ----
    k_feats<<<2048, 256, 0, stream>>>(hf_all, hb_all, linw16, lin_b, feats);

    // ---- CRF NLL + deterministic mean ----
    k_crf<<<8, 32, 0, stream>>>(feats, labels, transitions, nll_pb);
    k_reduce<<<1, BATCH, 0, stream>>>(nll_pb, out);
}
